// SelfModifyingMemory_18769007084153
// MI455X (gfx1250) — matrix operand order repaired and checked by reading, not compiled, not run on hardware
//
#include <hip/hip_runtime.h>
#include <hip/hip_bf16.h>

// ---------------------------------------------------------------------------
// SelfModifyingMemory for MI455X (gfx1250, wave32)
//   Phase 1: bf16 WMMA GEMMs (v_wmma_f32_16x16x32_bf16), 64x128 block tile,
//            2x2 subtiles per wave (4 WMMAs/K-step), register double-buffered
//            LDS staging with packed f32->2xbf16 conversion (single ds_store_b32
//            per pair), fused bias (+ReLU) epilogue.
//   Phase 2: row-parallel sequential scan, one wave per (batch,row); k_t is
//            double-buffered in LDS via GLOBAL_LOAD_ASYNC_TO_LDS_B64 with
//            s_wait_asynccnt pipelining (CDNA5 async-copy path).
// ---------------------------------------------------------------------------

typedef __attribute__((ext_vector_type(16))) __bf16 v16bf;
typedef __attribute__((ext_vector_type(8)))  float  v8f;

union BfFrag { v16bf v; unsigned short s[16]; };

static __device__ __forceinline__ unsigned short f32_to_bf16_bits(float f) {
    unsigned int u = __float_as_uint(f);
    unsigned int r = u + 0x7FFFu + ((u >> 16) & 1u);   // round-to-nearest-even
    return (unsigned short)(r >> 16);
}

// pack two f32 into one dword of 2x bf16 (lo, hi)
static __device__ __forceinline__ unsigned pack2_bf16(float lo, float hi) {
#if defined(__has_builtin)
#if __has_builtin(__builtin_amdgcn_cvt_pk_bf16_f32)
    typedef __attribute__((ext_vector_type(2))) __bf16 v2bf;
    union { v2bf v; unsigned u; } c;
    c.v = __builtin_amdgcn_cvt_pk_bf16_f32(lo, hi);
    return c.u;
#else
    return (unsigned)f32_to_bf16_bits(lo) | ((unsigned)f32_to_bf16_bits(hi) << 16);
#endif
#else
    return (unsigned)f32_to_bf16_bits(lo) | ((unsigned)f32_to_bf16_bits(hi) << 16);
#endif
}

#define TM 64
#define TN 128
#define TK 32
#define LDK (TK + 8)   // row stride in ushorts (80 B, 4B-aligned pair slots)

// C[M,N] = act(A[M,K] @ B[K,N] + bias[N]); act: 0=none, 1=relu
__global__ __launch_bounds__(256)
void wmma_gemm_bias(const float* __restrict__ A, const float* __restrict__ B,
                    const float* __restrict__ bias, float* __restrict__ C,
                    int M, int N, int K, int act) {
    __shared__ unsigned short As[2][TM][LDK];   // [m][k], bf16 bits
    __shared__ unsigned short Bs[2][TN][LDK];   // [n][k], bf16 bits (transposed)

    const int tid  = threadIdx.x;
    const int lane = tid & 31;
    const int wave = tid >> 5;
    const int half = lane >> 4;        // lane group (0: lanes 0-15, 1: 16-31)
    const int l15  = lane & 15;
    const int m0   = blockIdx.y * TM;
    const int n0   = blockIdx.x * TN;
    const int wm   = (wave & 1) * 32;  // wave's 32-row slot (2 subtiles in M)
    const int wn   = (wave >> 1) * 32; // wave's 32-col slot (2 subtiles in N)

    v8f acc00 = {}, acc01 = {}, acc10 = {}, acc11 = {};

    // Staging in K-pairs: A = 64x16 pairs (4/thread), B = 128x16 pairs (8/thread)
    float2 aReg[4];
    float  bLo[8], bHi[8];

    // ---- preload K-slab 0 into LDS buffer 0 ----
    #pragma unroll
    for (int i = 0; i < 4; ++i) {
        int idx = tid + i * 256; int r = idx >> 4, cp = idx & 15;
        aReg[i] = *(const float2*)&A[(size_t)(m0 + r) * K + 2 * cp];
    }
    #pragma unroll
    for (int i = 0; i < 8; ++i) {
        int idx = tid + i * 256; int nn = idx & 127, kp = idx >> 7;
        bLo[i] = B[(size_t)(2 * kp) * N + (n0 + nn)];
        bHi[i] = B[(size_t)(2 * kp + 1) * N + (n0 + nn)];
    }
    #pragma unroll
    for (int i = 0; i < 4; ++i) {
        int idx = tid + i * 256; int r = idx >> 4, cp = idx & 15;
        *(unsigned*)&As[0][r][2 * cp] = pack2_bf16(aReg[i].x, aReg[i].y);
    }
    #pragma unroll
    for (int i = 0; i < 8; ++i) {
        int idx = tid + i * 256; int nn = idx & 127, kp = idx >> 7;
        *(unsigned*)&Bs[0][nn][2 * kp] = pack2_bf16(bLo[i], bHi[i]);
    }
    __syncthreads();

    const int steps = K / TK;
    for (int st = 0; st < steps; ++st) {
        const int buf = st & 1;
        const int kn  = (st + 1) * TK;            // next slab base

        if (st + 2 < steps) {                     // L2 prefetch hints (+2 slabs)
            __builtin_prefetch(&A[(size_t)(m0 + (tid >> 3)) * K + (kn + TK)], 0, 1);
            __builtin_prefetch(&B[(size_t)(kn + TK + (tid >> 3)) * N + n0 + 4 * (tid & 7)], 0, 1);
        }
        if (st + 1 < steps) {                     // global -> regs for next slab
            #pragma unroll
            for (int i = 0; i < 4; ++i) {
                int idx = tid + i * 256; int r = idx >> 4, cp = idx & 15;
                aReg[i] = *(const float2*)&A[(size_t)(m0 + r) * K + (kn + 2 * cp)];
            }
            #pragma unroll
            for (int i = 0; i < 8; ++i) {
                int idx = tid + i * 256; int nn = idx & 127, kp = idx >> 7;
                bLo[i] = B[(size_t)(kn + 2 * kp) * N + (n0 + nn)];
                bHi[i] = B[(size_t)(kn + 2 * kp + 1) * N + (n0 + nn)];
            }
        }

        // ---- fragments ----
        // A, 16-bit 16x32 layout: lanes 0-15: elems 0-7 -> K 0-7, 8-15 -> K 16-23
        //                         lanes 16-31: elems 0-7 -> K 8-15, 8-15 -> K 24-31
        BfFrag a0, a1;
        #pragma unroll
        for (int e = 0; e < 8; ++e) {
            a0.s[e]     = As[buf][wm + l15][half * 8 + e];
            a0.s[8 + e] = As[buf][wm + l15][16 + half * 8 + e];
            a1.s[e]     = As[buf][wm + 16 + l15][half * 8 + e];
            a1.s[8 + e] = As[buf][wm + 16 + l15][16 + half * 8 + e];
        }
        // B: lane holds column n=l15; same K order as A (elems 0-7 -> K 8*half+e, 8-15 -> K 16+8*half+e)
        BfFrag b0, b1;
        #pragma unroll
        for (int e = 0; e < 8; ++e) {
            b0.s[e]     = Bs[buf][wn + l15][half * 8 + e];
            b0.s[8 + e] = Bs[buf][wn + l15][16 + half * 8 + e];
            b1.s[e]     = Bs[buf][wn + 16 + l15][half * 8 + e];
            b1.s[8 + e] = Bs[buf][wn + 16 + l15][16 + half * 8 + e];
        }
        acc00 = __builtin_amdgcn_wmma_f32_16x16x32_bf16(false, a0.v, false, b0.v,
                                                        (short)0, acc00, false, false);
        acc01 = __builtin_amdgcn_wmma_f32_16x16x32_bf16(false, a0.v, false, b1.v,
                                                        (short)0, acc01, false, false);
        acc10 = __builtin_amdgcn_wmma_f32_16x16x32_bf16(false, a1.v, false, b0.v,
                                                        (short)0, acc10, false, false);
        acc11 = __builtin_amdgcn_wmma_f32_16x16x32_bf16(false, a1.v, false, b1.v,
                                                        (short)0, acc11, false, false);

        if (st + 1 < steps) {                     // regs -> LDS (other buffer)
            #pragma unroll
            for (int i = 0; i < 4; ++i) {
                int idx = tid + i * 256; int r = idx >> 4, cp = idx & 15;
                *(unsigned*)&As[buf ^ 1][r][2 * cp] = pack2_bf16(aReg[i].x, aReg[i].y);
            }
            #pragma unroll
            for (int i = 0; i < 8; ++i) {
                int idx = tid + i * 256; int nn = idx & 127, kp = idx >> 7;
                *(unsigned*)&Bs[buf ^ 1][nn][2 * kp] = pack2_bf16(bLo[i], bHi[i]);
            }
        }
        __syncthreads();
    }

    // ---- epilogue: C/D layout -> VGPR r holds M = r + 8*half, N = l15 ----
    #pragma unroll
    for (int r = 0; r < 8; ++r) {
        const int mA = m0 + wm + r + 8 * half;
        const int mB = mA + 16;
        const int nA = n0 + wn + l15;
        const int nB = nA + 16;
        float v00 = acc00[r] + bias[nA];
        float v01 = acc01[r] + bias[nB];
        float v10 = acc10[r] + bias[nA];
        float v11 = acc11[r] + bias[nB];
        if (act == 1) {
            v00 = fmaxf(v00, 0.0f); v01 = fmaxf(v01, 0.0f);
            v10 = fmaxf(v10, 0.0f); v11 = fmaxf(v11, 0.0f);
        }
        C[(size_t)mA * N + nA] = v00;
        C[(size_t)mA * N + nB] = v01;
        C[(size_t)mB * N + nA] = v10;
        C[(size_t)mB * N + nB] = v11;
    }
}

// out[t] = sigmoid(dot(H[t,:], w) + b)   (H: [T,Hd], w: [Hd])
__global__ __launch_bounds__(256)
void rowdot_sigmoid(const float* __restrict__ H, const float* __restrict__ w,
                    const float* __restrict__ b, float* __restrict__ out, int Hd) {
    __shared__ float red[256];
    const int t = blockIdx.x;
    float p = 0.f;
    for (int i = threadIdx.x; i < Hd; i += 256)
        p += H[(size_t)t * Hd + i] * w[i];
    red[threadIdx.x] = p;
    __syncthreads();
    for (int s = 128; s > 0; s >>= 1) {
        if (threadIdx.x < s) red[threadIdx.x] += red[threadIdx.x + s];
        __syncthreads();
    }
    if (threadIdx.x == 0) {
        float z = red[0] + b[0];
        out[t] = 1.0f / (1.0f + __expf(-z));
    }
}

// decay[t] = alpha[t] * (1 - eta[t] * ||k[t,:]||^2)
__global__ __launch_bounds__(256)
void decay_kernel(const float* __restrict__ Kb, const float* __restrict__ eta,
                  const float* __restrict__ alpha, float* __restrict__ decay, int D) {
    __shared__ float red[256];
    const int t = blockIdx.x;
    float p = 0.f;
    for (int i = threadIdx.x; i < D; i += 256) {
        float kv = Kb[(size_t)t * D + i];
        p += kv * kv;
    }
    red[threadIdx.x] = p;
    __syncthreads();
    for (int s = 128; s > 0; s >>= 1) {
        if (threadIdx.x < s) red[threadIdx.x] += red[threadIdx.x + s];
        __syncthreads();
    }
    if (threadIdx.x == 0)
        decay[t] = alpha[t] * (1.0f - eta[t] * red[0]);
}

// Sequential scan over S steps; one wave per (batch,row), 8 rows per block.
// k_t double-buffered in LDS via CDNA5 async global->LDS copies.
#define SCAN_ROWS 8
__global__ __launch_bounds__(256)
void scan_kernel(const float* __restrict__ Kb, const float* __restrict__ vhk,
                 const float* __restrict__ eta, const float* __restrict__ decay,
                 const float* __restrict__ init_mem, float* __restrict__ mem_out,
                 int B, int S, int D) {
    __shared__ float kbuf[2][512];
    const int rowsPerB = D / SCAN_ROWS;
    const int b    = blockIdx.x / rowsPerB;
    const int rb   = blockIdx.x % rowsPerB;
    const int lane = threadIdx.x & 31;
    const int wave = threadIdx.x >> 5;
    const int row  = rb * SCAN_ROWS + wave;
    const int col2 = threadIdx.x * 2;            // this thread's 8-byte chunk
    const size_t kbase = (size_t)b * S * D;

    float m[16];  // 512 cols / 32 lanes
    #pragma unroll
    for (int j = 0; j < 16; ++j)
        m[j] = init_mem[(size_t)row * D + (lane + 32 * j)];

    // prologue: async-copy k_0 into kbuf[0] (ASYNCcnt = 1)
    {
        unsigned lds = (unsigned)(size_t)&kbuf[0][col2];
        unsigned long long ga =
            (unsigned long long)(size_t)(Kb + kbase + col2);
        asm volatile("global_load_async_to_lds_b64 %0, %1, off"
                     :: "v"(lds), "v"(ga) : "memory");
    }

    for (int t = 0; t < S; ++t) {
        const int buf = t & 1;
        if (t + 1 < S) {
            // issue async copy of k_{t+1} into the other buffer, then wait
            // until only that newest copy is outstanding (async loads retire
            // in order -> k_t has fully landed in LDS).
            unsigned lds = (unsigned)(size_t)&kbuf[buf ^ 1][col2];
            unsigned long long ga =
                (unsigned long long)(size_t)(Kb + kbase + (size_t)(t + 1) * D + col2);
            asm volatile("global_load_async_to_lds_b64 %0, %1, off"
                         :: "v"(lds), "v"(ga) : "memory");
            asm volatile("s_wait_asynccnt 0x1" ::: "memory");
        } else {
            asm volatile("s_wait_asynccnt 0x0" ::: "memory");
        }
        __syncthreads();   // every thread's chunk of k_t is visible

        const float dc = decay[b * S + t];
        const float et = eta[b * S + t];
        const float vh = vhk[(kbase + (size_t)t * D) + row];

        float p = 0.f;
        #pragma unroll
        for (int j = 0; j < 16; ++j) {
            m[j] *= dc;                            // mem = mem * decay
            p += m[j] * kbuf[buf][lane + 32 * j];  // mk_row = dot(mem_row, k_t)
        }
        #pragma unroll
        for (int off = 16; off >= 1; off >>= 1)
            p += __shfl_xor(p, off, 32);

        const float coef = et * (p - vh);
        #pragma unroll
        for (int j = 0; j < 16; ++j)
            m[j] -= coef * kbuf[buf][lane + 32 * j];

        __syncthreads();   // all reads of kbuf[buf] done before t+1 reuses it
    }

    #pragma unroll
    for (int j = 0; j < 16; ++j)
        mem_out[((size_t)b * D + row) * D + (lane + 32 * j)] = m[j];
}

// new_mem_v = broadcast(init_mem_v) over B
__global__ __launch_bounds__(256)
void bcast_kernel(const float* __restrict__ src, float* __restrict__ dst,
                  int DD, size_t total) {
    size_t i = (size_t)blockIdx.x * 256 + threadIdx.x;
    if (i < total) dst[i] = src[i & (size_t)(DD - 1)];
}

extern "C" void kernel_launch(void* const* d_in, const int* in_sizes, int n_in,
                              void* d_out, int out_size, void* d_ws, size_t ws_size,
                              hipStream_t stream) {
    const int B = 4, S = 2048, D = 512, Hh = 2048;
    const size_t BS = (size_t)B * S;          // 8192 tokens

    const float* x          = (const float*)d_in[0];
    const float* mk_w1      = (const float*)d_in[1];
    const float* mk_b1      = (const float*)d_in[2];
    const float* mk_w2      = (const float*)d_in[3];
    const float* mk_b2      = (const float*)d_in[4];
    const float* mv_w1      = (const float*)d_in[5];
    const float* mv_b1      = (const float*)d_in[6];
    const float* mv_w2      = (const float*)d_in[7];
    const float* mv_b2      = (const float*)d_in[8];
    const float* mq_w1      = (const float*)d_in[9];
    const float* mq_b1      = (const float*)d_in[10];
    const float* mq_w2      = (const float*)d_in[11];
    const float* mq_b2      = (const float*)d_in[12];
    const float* mo_w1      = (const float*)d_in[13];
    const float* mo_b1      = (const float*)d_in[14];
    const float* mo_w2      = (const float*)d_in[15];
    const float* mo_b2      = (const float*)d_in[16];
    const float* eta_w1     = (const float*)d_in[17];
    const float* eta_b1     = (const float*)d_in[18];
    const float* eta_w2     = (const float*)d_in[19];
    const float* eta_b2     = (const float*)d_in[20];
    const float* alp_w1     = (const float*)d_in[21];
    const float* alp_b1     = (const float*)d_in[22];
    const float* alp_w2     = (const float*)d_in[23];
    const float* alp_b2     = (const float*)d_in[24];
    const float* vgk_w      = (const float*)d_in[25];
    const float* vgk_b      = (const float*)d_in[26];
    const float* init_mem_k = (const float*)d_in[27];
    const float* init_mem_v = (const float*)d_in[28];

    // Workspace carve-up (~135 MB)
    float* ws   = (float*)d_ws;
    float* hbuf = ws;                  // BS * Hh  (reused across all first layers)
    float* kbuf = hbuf + BS * Hh;      // BS * D
    float* vbuf = kbuf + BS * D;       // BS * D
    float* qbuf = vbuf + BS * D;       // BS * D
    float* vhk  = qbuf + BS * D;       // BS * D
    float* etab = vhk  + BS * D;       // BS
    float* alpb = etab + BS;           // BS
    float* decb = alpb + BS;           // BS

    float* out_y    = (float*)d_out;                 // (B,S,D)
    float* out_memk = out_y + BS * D;                // (B,D,D)
    float* out_memv = out_memk + (size_t)B * D * D;  // (B,D,D)

    dim3 blk(256);
    auto gemm = [&](const float* A, const float* Bm, const float* bias, float* C,
                    int M, int N, int K, int act) {
        dim3 grid(N / TN, M / TM);
        wmma_gemm_bias<<<grid, blk, 0, stream>>>(A, Bm, bias, C, M, N, K, act);
    };

    // k = relu(x@w1+b1)@w2+b2 ; same for v, q
    gemm(x, mk_w1, mk_b1, hbuf, (int)BS, Hh, D, 1);
    gemm(hbuf, mk_w2, mk_b2, kbuf, (int)BS, D, Hh, 0);
    gemm(x, mv_w1, mv_b1, hbuf, (int)BS, Hh, D, 1);
    gemm(hbuf, mv_w2, mv_b2, vbuf, (int)BS, D, Hh, 0);
    gemm(x, mq_w1, mq_b1, hbuf, (int)BS, Hh, D, 1);
    gemm(hbuf, mq_w2, mq_b2, qbuf, (int)BS, D, Hh, 0);

    // eta / alpha heads (H -> 1, sigmoid)
    gemm(x, eta_w1, eta_b1, hbuf, (int)BS, Hh, D, 1);
    rowdot_sigmoid<<<dim3((int)BS), blk, 0, stream>>>(hbuf, eta_w2, eta_b2, etab, Hh);
    gemm(x, alp_w1, alp_b1, hbuf, (int)BS, Hh, D, 1);
    rowdot_sigmoid<<<dim3((int)BS), blk, 0, stream>>>(hbuf, alp_w2, alp_b2, alpb, Hh);

    // v_hat_k = v @ vgk_w + vgk_b
    gemm(vbuf, vgk_w, vgk_b, vhk, (int)BS, D, D, 0);

    // per-step decay scalars
    decay_kernel<<<dim3((int)BS), blk, 0, stream>>>(kbuf, etab, alpb, decb, D);

    // sequential memory scan -> new_mem_k
    scan_kernel<<<dim3(B * (D / SCAN_ROWS)), blk, 0, stream>>>(
        kbuf, vhk, etab, decb, init_mem_k, out_memk, B, S, D);

    // output = relu(q@w1+b1)@w2+b2
    gemm(qbuf, mo_w1, mo_b1, hbuf, (int)BS, Hh, D, 1);
    gemm(hbuf, mo_w2, mo_b2, out_y, (int)BS, D, Hh, 0);

    // new_mem_v = broadcast(init_mem_v)
    size_t totalV = (size_t)B * D * D;
    bcast_kernel<<<dim3((int)((totalV + 255) / 256)), blk, 0, stream>>>(
        init_mem_v, out_memv, D * D, totalV);
}
